// Attention_13503377178994
// MI455X (gfx1250) — matrix operand order repaired and checked by reading, not compiled, not run on hardware
//
#include <hip/hip_runtime.h>
#include <hip/hip_bf16.h>

// ---------------------------------------------------------------------------
// Attention: out = softmax((hostQ)(guestK)^T / 32) @ (guestV)
// N=4096 queries, M=8192 keys, d=1024. ~180 GFLOP, working set ~66MB (L2-
// resident on MI455X's 192MB L2) => compute bound => everything on
// v_wmma_f32_16x16x32_bf16 (bf16 inputs, f32 accumulate), flash-attention
// style (never materialize the 4096x8192 score matrix).
// ---------------------------------------------------------------------------

typedef __attribute__((ext_vector_type(16))) __bf16 v16bf;
typedef __attribute__((ext_vector_type(8)))  __bf16 v8bf;
typedef __attribute__((ext_vector_type(8)))  float  v8f;

#define DIMD 1024
#define NH   4096
#define NG   8192
#define BR   32
#define BC   64

union FragA { v16bf v; v8bf h[2]; };

// ---------------------------------------------------------------------------
// Kernel 1: C = X(f32)[N x 1024] @ W(f32)[1024 x 1024], output bf16.
// transpose==0 : out[n*1024 + j]   (row-major, used for query/key)
// transpose==1 : out[j*N + n]      (d-major,  used for value^T)
// Block: 256 threads = 8 waves; wave w owns 16x16 tile at rows blk.x*128+16w,
// cols blk.y*16. The 32x16 W tile (shared by all 8 waves) is staged in LDS
// transposed (Wt[n][k]) so the B fragment is one contiguous 32B read/lane.
// f32->bf16 uses native converts (vector fptrunc), not bit-twiddling.
// ---------------------------------------------------------------------------
__launch_bounds__(256)
__global__ void gemm_bf16(const float* __restrict__ X, const float* __restrict__ W,
                          __bf16* __restrict__ out, int N, int transpose)
{
  __shared__ __attribute__((aligned(32))) __bf16 Wt[16][48]; // padded rows (96B, 32B-aligned)

  const int tid  = threadIdx.x;
  const int lane = tid & 31;
  const int wave = tid >> 5;
  const int half = lane >> 4;
  const int l16  = lane & 15;

  const int row0 = blockIdx.x * 128 + wave * 16;
  const int col0 = blockIdx.y * 16;

  v8f c = {};
  const float* xr = X + (size_t)(row0 + l16) * DIMD;

  for (int kb = 0; kb < DIMD; kb += 32) {
    __syncthreads();
    // cooperative stage of W[kb..kb+31][col0..col0+15] -> Wt[n][k] (bf16)
    for (int e = tid; e < 512; e += 256) {
      int k = e >> 4, n = e & 15;
      Wt[n][k] = (__bf16)W[(size_t)(kb + k) * DIMD + col0 + n];
    }
    __syncthreads();

    // A fragment: lane row = l16; 16 contiguous K at 16*half
    FragA a;
    a.h[0] = __builtin_convertvector(*(const v8f*)(xr + kb + 16 * half), v8bf);
    a.h[1] = __builtin_convertvector(*(const v8f*)(xr + kb + 16 * half + 8), v8bf);

    // B fragment: lane col = l16, 16 contiguous K at 16*half
    v16bf b = *(const v16bf*)(&Wt[l16][16 * half]);

    c = __builtin_amdgcn_wmma_f32_16x16x32_bf16(false, a.v, false, b, (short)0, c, false, false);
  }

  // D layout: VGPR r -> row = r (+8 for lanes 16..31), col = l16
  v8bf cb = __builtin_convertvector(c, v8bf);
  for (int r = 0; r < 8; ++r) {
    int row = row0 + r + 8 * half;
    int col = col0 + l16;
    if (!transpose) out[(size_t)row * DIMD + col] = cb[r];
    else            out[(size_t)col * N    + row] = cb[r];
  }
}

// ---------------------------------------------------------------------------
// Kernel 2: flash attention. One block per 32 query rows, 8 waves.
// Phase 1 role (wr1=w>>2, wc1=w&3): wave computes 16x16 tile of S = QK^T/32.
// Phase 2 role (wr2=w&1, wd2=w>>1): wave accumulates O for 16 rows x 256 d.
// Online softmax stats combined across the 4 column-waves through LDS.
// Next key/V tiles prefetched (global_prefetch) while current tile computes.
// ---------------------------------------------------------------------------
__launch_bounds__(256)
__global__ void flash_attn(const __bf16* __restrict__ qB,  // [NH][D] bf16
                           const __bf16* __restrict__ kB,  // [NG][D] bf16
                           const __bf16* __restrict__ vT,  // [D][NG] bf16
                           float* __restrict__ out)        // [NH][D] f32
{
  __shared__ __attribute__((aligned(32))) __bf16 Qs[BR][DIMD];   // 64KB
  __shared__ __attribute__((aligned(32))) __bf16 Ps[2][16][BC];  // 4KB
  __shared__ float smax[2][16][4];
  __shared__ float ssum[2][16][4];
  __shared__ float mS[BR], lS[BR], aS[BR], mnS[BR];

  const int tid  = threadIdx.x;
  const int lane = tid & 31;
  const int wave = tid >> 5;
  const int half = lane >> 4;
  const int l16  = lane & 15;
  const int qrow0 = blockIdx.x * BR;

  // Stage the query block (bf16) into LDS: reused across all 128 key tiles.
  {
    const uint4* src = (const uint4*)(qB + (size_t)qrow0 * DIMD);
    uint4* dst = (uint4*)&Qs[0][0];
    for (int i = tid; i < BR * DIMD / 8; i += 256) dst[i] = src[i];
  }
  if (tid < BR) { mS[tid] = -__builtin_inff(); lS[tid] = 0.f; }
  __syncthreads();

  const int wr1 = wave >> 2, wc1 = wave & 3;   // score-tile role
  const int wr2 = wave & 1,  wd2 = wave >> 1;  // output-slice role (d base = wd2*256)

  v8f oacc[16] = {};                            // 16 rows x 256 d, f32
  const float scale = 0.03125f;                 // 1/sqrt(1024)

  for (int jt = 0; jt < NG / BC; ++jt) {
    const int mbase = jt * BC;

    // prefetch next iteration's tiles toward this WGP (L2-resident data)
    if (jt + 1 < NG / BC) {
      __builtin_prefetch(kB + (size_t)(mbase + BC + wc1 * 16 + l16) * DIMD, 0, 3);
      __builtin_prefetch(vT + (size_t)(wd2 * 256 + wave * 2 + half) * NG + mbase + BC, 0, 3);
    }

    // ---------------- Phase 1: S tile (16x16) = Q Kt over d=1024 ----------
    v8f c = {};
    const __bf16* qrow = &Qs[wr1 * 16 + l16][0];
    const __bf16* krow = kB + (size_t)(mbase + wc1 * 16 + l16) * DIMD;
    for (int kd = 0; kd < DIMD; kd += 32) {
      FragA a;
      a.h[0] = *(const v8bf*)(qrow + kd + 16 * half);
      a.h[1] = *(const v8bf*)(qrow + kd + 16 * half + 8);
      v16bf b = *(const v16bf*)(krow + kd + 16 * half);
      c = __builtin_amdgcn_wmma_f32_16x16x32_bf16(false, a.v, false, b, (short)0, c, false, false);
    }
    for (int r = 0; r < 8; ++r) c[r] *= scale;

    // per-row max over this wave's 16 columns (reduce across 16-lane group)
    for (int r = 0; r < 8; ++r) {
      float v = c[r];
      v = fmaxf(v, __shfl_xor(v, 1, 32));
      v = fmaxf(v, __shfl_xor(v, 2, 32));
      v = fmaxf(v, __shfl_xor(v, 4, 32));
      v = fmaxf(v, __shfl_xor(v, 8, 32));
      if (l16 == 0) smax[wr1][r + 8 * half][wc1] = v;
    }
    __syncthreads();

    // one thread per query row: combine 4 partial maxima, online-max update
    if (tid < BR) {
      int rt = tid >> 4, rr = tid & 15;
      float tm = fmaxf(fmaxf(smax[rt][rr][0], smax[rt][rr][1]),
                       fmaxf(smax[rt][rr][2], smax[rt][rr][3]));
      float mo = mS[tid];
      float mn = fmaxf(mo, tm);
      mS[tid]  = mn;
      mnS[tid] = mn;
      aS[tid]  = __expf(mo - mn);               // rescale factor for O, l
    }
    __syncthreads();

    // P = exp(S - m_new): write bf16 to LDS, accumulate partial row sums
    for (int r = 0; r < 8; ++r) {
      float mn = mnS[wr1 * 16 + r + 8 * half];
      float p = __expf(c[r] - mn);
      Ps[wr1][r + 8 * half][wc1 * 16 + l16] = (__bf16)p;
      float v = p;
      v += __shfl_xor(v, 1, 32);
      v += __shfl_xor(v, 2, 32);
      v += __shfl_xor(v, 4, 32);
      v += __shfl_xor(v, 8, 32);
      if (l16 == 0) ssum[wr1][r + 8 * half][wc1] = v;
    }
    __syncthreads();

    if (tid < BR) {
      int rt = tid >> 4, rr = tid & 15;
      float ts = ssum[rt][rr][0] + ssum[rt][rr][1] + ssum[rt][rr][2] + ssum[rt][rr][3];
      lS[tid] = lS[tid] * aS[tid] + ts;
    }

    // ---------------- Phase 2: O = O*alpha + P @ V ------------------------
    float al[8];
    for (int r = 0; r < 8; ++r) al[r] = aS[wr2 * 16 + r + 8 * half];
    for (int t = 0; t < 16; ++t)
      for (int r = 0; r < 8; ++r) oacc[t][r] *= al[r];

    for (int km = 0; km < BC; km += 32) {
      FragA a;
      const __bf16* prow = &Ps[wr2][l16][0];
      a.h[0] = *(const v8bf*)(prow + km + 16 * half);
      a.h[1] = *(const v8bf*)(prow + km + 16 * half + 8);
      for (int t = 0; t < 16; ++t) {
        int dd = wd2 * 256 + t * 16 + l16;
        v16bf b = *(const v16bf*)(vT + (size_t)dd * NG + mbase + km + 16 * half);
        oacc[t] = __builtin_amdgcn_wmma_f32_16x16x32_bf16(false, a.v, false, b, (short)0, oacc[t], false, false);
      }
    }
    __syncthreads();
  }

  // epilogue: normalize by l and store f32
  float linv[8];
  for (int r = 0; r < 8; ++r) linv[r] = 1.f / lS[wr2 * 16 + r + 8 * half];
  for (int t = 0; t < 16; ++t) {
    for (int r = 0; r < 8; ++r) {
      int row = qrow0 + wr2 * 16 + r + 8 * half;
      int dd  = wd2 * 256 + t * 16 + l16;
      out[(size_t)row * DIMD + dd] = oacc[t][r] * linv[r];
    }
  }
}

// ---------------------------------------------------------------------------
extern "C" void kernel_launch(void* const* d_in, const int* in_sizes, int n_in,
                              void* d_out, int out_size, void* d_ws, size_t ws_size,
                              hipStream_t stream) {
  const float* host   = (const float*)d_in[0];   // [4096,1024]
  const float* guests = (const float*)d_in[1];   // [8192,1024]
  const float* Qw     = (const float*)d_in[2];   // [1024,1024]
  const float* Kw     = (const float*)d_in[3];
  const float* Vw     = (const float*)d_in[4];
  float* out = (float*)d_out;                    // [4096,1024]

  // workspace (bf16): query 8MB | key 16MB | value^T 16MB  (40MB total)
  __bf16* qB  = (__bf16*)d_ws;
  __bf16* kBf = qB  + (size_t)NH * DIMD;
  __bf16* vT  = kBf + (size_t)NG * DIMD;

  gemm_bf16<<<dim3(NH / 128, DIMD / 16), 256, 0, stream>>>(host,   Qw, qB,  NH, 0);
  gemm_bf16<<<dim3(NG / 128, DIMD / 16), 256, 0, stream>>>(guests, Kw, kBf, NG, 0);
  gemm_bf16<<<dim3(NG / 128, DIMD / 16), 256, 0, stream>>>(guests, Vw, vT,  NG, 1);
  flash_attn<<<dim3(NH / BR), 256, 0, stream>>>(qB, kBf, vT, out);
}
